// TransformerDecoderLayer_56324201120062
// MI455X (gfx1250) — matrix operand order repaired and checked by reading, not compiled, not run on hardware
//
#include <hip/hip_runtime.h>
#include <hip/hip_bf16.h>

#define BB   32
#define SS   1600
#define DD   128
#define DFF  2048
#define ST   100          // S / 16 tiles
#define RS_BN 0.99999500003749972f  // 1/sqrt(1+1e-5)

typedef __attribute__((ext_vector_type(8)))  float  v8f;
typedef __attribute__((ext_vector_type(16))) __bf16 v16bf;

union Frag { uint4 q[2]; unsigned u[8]; v16bf v; };

__device__ __forceinline__ unsigned bfpack(float x, float y) {
    unsigned a = __builtin_bit_cast(unsigned, x);
    unsigned b = __builtin_bit_cast(unsigned, y);
    a = (a + 0x7fffu + ((a >> 16) & 1u)) >> 16;
    b = (b + 0x7fffu + ((b >> 16) & 1u)) >> 16;
    return a | (b << 16);
}
__device__ __forceinline__ unsigned short bf1(float x) {
    unsigned a = __builtin_bit_cast(unsigned, x);
    return (unsigned short)((a + 0x7fffu + ((a >> 16) & 1u)) >> 16);
}

// ---------------------------------------------------------------------------
// Kernel 1: q = l2norm(x @ w1^T + b1) for tgt and memory.
// Output: Q in WMMA A-fragment layout, K in WMMA B-fragment layout (bf16).
// grid 1600 x 128 threads (4 waves, one 16-row tile per wave; 6400 tiles).
// ---------------------------------------------------------------------------
__global__ void qk_kernel(const float* __restrict__ tgt,
                          const float* __restrict__ mem,
                          const float* __restrict__ w1,
                          const float* __restrict__ b1,
                          unsigned* __restrict__ Qf,
                          unsigned* __restrict__ Kf) {
    __shared__ unsigned w1lds[128 * 64];      // 128x128 bf16 pairs (32 KB)
    __shared__ unsigned atile[4][16 * 64];    // per-wave 16x128 bf16 (4x4 KB)

    const int tid  = threadIdx.x;
    const int wave = tid >> 5, lane = tid & 31;
    const int m = lane & 15, hf = lane >> 4;

    // stage w1 as bf16 pairs
    for (int i = tid; i < 8192; i += 128) {
        float2 p = ((const float2*)w1)[i];
        w1lds[i] = bfpack(p.x, p.y);
    }

    const int tile = blockIdx.x * 4 + wave;        // 0..6399
    const bool isQ = tile < 3200;
    const int  rt  = isQ ? tile : tile - 3200;     // row tile 0..3199
    const float* src = isQ ? tgt : mem;
    unsigned*    dst = isQ ? Qf  : Kf;
    const long rowbase = (long)rt * 16;

    // stage 16x128 input rows as bf16 pairs
    unsigned* al = atile[wave];
    const float2* srow = (const float2*)(src + rowbase * DD);
    for (int i = lane; i < 1024; i += 32) {
        float2 p = srow[i];
        al[i] = bfpack(p.x, p.y);
    }
    __syncthreads();

    float bias[8];
#pragma unroll
    for (int nb = 0; nb < 8; ++nb) bias[nb] = b1[nb * 16 + m];

    // A fragments: 4 K-steps of 32
    unsigned afr[4][8];
#pragma unroll
    for (int kk = 0; kk < 4; ++kk)
#pragma unroll
        for (int v = 0; v < 8; ++v) {
            int kp = kk * 32 + hf * 8 + (v & 3) * 2 + (v >> 2) * 16;
            afr[kk][v] = al[m * 64 + (kp >> 1)];
        }

    v8f acc[8];
#pragma unroll
    for (int nb = 0; nb < 8; ++nb) {
        v8f a;
#pragma unroll
        for (int v = 0; v < 8; ++v) a[v] = bias[nb];
        const int n = nb * 16 + m;
#pragma unroll
        for (int kk = 0; kk < 4; ++kk) {
            Frag bfr;
#pragma unroll
            for (int v = 0; v < 8; ++v) {
                int kp = kk * 32 + hf * 8 + (v & 3) * 2 + (v >> 2) * 16;
                bfr.u[v] = w1lds[n * 64 + (kp >> 1)];
            }
            Frag av;
#pragma unroll
            for (int v = 0; v < 8; ++v) av.u[v] = afr[kk][v];
            a = __builtin_amdgcn_wmma_f32_16x16x32_bf16(
                    false, av.v, false, bfr.v, (short)0, a, false, false);
        }
        acc[nb] = a;
    }

    // per-row sum of squares -> reduce over the 16 N-lanes -> normalize
    float ss[8];
#pragma unroll
    for (int v = 0; v < 8; ++v) {
        float s = 0.f;
#pragma unroll
        for (int nb = 0; nb < 8; ++nb) s += acc[nb][v] * acc[nb][v];
#pragma unroll
        for (int msk = 1; msk < 16; msk <<= 1) s += __shfl_xor(s, msk, 32);
        ss[v] = 1.0f / fmaxf(sqrtf(s), 1e-12f);
    }

    __syncthreads();
    // write normalized rows (row-major bf16) back to LDS, then emit fragments
    unsigned short* as16 = (unsigned short*)al;
#pragma unroll
    for (int nb = 0; nb < 8; ++nb)
#pragma unroll
        for (int v = 0; v < 8; ++v)
            as16[(v + 8 * hf) * 128 + nb * 16 + m] = bf1(acc[nb][v] * ss[v]);
    __syncthreads();

#pragma unroll
    for (int kk = 0; kk < 4; ++kk) {
        unsigned e[8];
        if (isQ) {
#pragma unroll
            for (int v = 0; v < 8; ++v) {
                int kp = kk * 32 + hf * 8 + (v & 3) * 2 + (v >> 2) * 16;
                e[v] = al[m * 64 + (kp >> 1)];
            }
        } else {
#pragma unroll
            for (int v = 0; v < 8; ++v) {
                int kp = kk * 32 + hf * 8 + (v & 3) * 2 + (v >> 2) * 16;
                e[v] = al[m * 64 + (kp >> 1)];
            }
        }
        uint4* o = (uint4*)&dst[(((rt * 4 + kk) * 32) + lane) * 8];
        o[0] = make_uint4(e[0], e[1], e[2], e[3]);
        o[1] = make_uint4(e[4], e[5], e[6], e[7]);
    }
}

// ---------------------------------------------------------------------------
// gate = sigmoid(score_embed)
// ---------------------------------------------------------------------------
__global__ void gate_kernel(const float* __restrict__ se, float* __restrict__ g) {
    int i = blockIdx.x * 256 + threadIdx.x;
    if (i < SS * SS) g[i] = 1.0f / (1.0f + __expf(-se[i]));
}

// ---------------------------------------------------------------------------
// Kernel 2: attn = (Q K^T) * gate; fused row-max (m2) + per-s_tile col-max.
// grid (50, 32) x 128 threads; each workgroup covers TWO s-tiles (32 rows),
// each wave handles 25 t-tiles -> 8 WMMA per K-fragment load (2 indep chains).
// ---------------------------------------------------------------------------
__global__ void attn_kernel(const unsigned* __restrict__ Qf,
                            const unsigned* __restrict__ Kf,
                            const float* __restrict__ gate,
                            float* __restrict__ m1part,
                            float* __restrict__ m2) {
    const int sp = blockIdx.x, b = blockIdx.y;     // s-pair, batch
    const int st0 = sp * 2;                        // first of two s-tiles
    const int tid = threadIdx.x;
    const int wave = tid >> 5, lane = tid & 31;
    const int m = lane & 15, hf = lane >> 4;

    Frag afr[2][4];
#pragma unroll
    for (int p = 0; p < 2; ++p) {
        const int qt = b * ST + st0 + p;
#pragma unroll
        for (int kk = 0; kk < 4; ++kk) {
            const uint4* ptr = (const uint4*)&Qf[(((qt * 4 + kk) * 32) + lane) * 8];
            afr[p][kk].q[0] = ptr[0];
            afr[p][kk].q[1] = ptr[1];
        }
    }

    float rmax[2][8];
#pragma unroll
    for (int p = 0; p < 2; ++p)
#pragma unroll
        for (int v = 0; v < 8; ++v) rmax[p][v] = -3.4e38f;

    for (int tt = wave; tt < ST; tt += 4) {
        const int kt = b * ST + tt;

        // prefetch next iteration's K tile (global_prefetch_b8)
        if (tt + 4 < ST) {
            const unsigned* nxt = &Kf[((((kt + 4) * 4) * 32) + lane) * 8];
#pragma unroll
            for (int kk = 0; kk < 4; ++kk)
                __builtin_prefetch(nxt + kk * 256, 0, 3);
        }

        v8f acc[2];
#pragma unroll
        for (int p = 0; p < 2; ++p)
#pragma unroll
            for (int v = 0; v < 8; ++v) acc[p][v] = 0.f;

#pragma unroll
        for (int kk = 0; kk < 4; ++kk) {
            Frag bfr;
            const uint4* ptr = (const uint4*)&Kf[(((kt * 4 + kk) * 32) + lane) * 8];
            bfr.q[0] = ptr[0];
            bfr.q[1] = ptr[1];
            acc[0] = __builtin_amdgcn_wmma_f32_16x16x32_bf16(
                         false, afr[0][kk].v, false, bfr.v, (short)0, acc[0], false, false);
            acc[1] = __builtin_amdgcn_wmma_f32_16x16x32_bf16(
                         false, afr[1][kk].v, false, bfr.v, (short)0, acc[1], false, false);
        }

#pragma unroll
        for (int p = 0; p < 2; ++p) {
            float cm = -3.4e38f;
#pragma unroll
            for (int v = 0; v < 8; ++v) {
                const int row = (st0 + p) * 16 + v + 8 * hf;
                const float a = acc[p][v] * gate[row * SS + tt * 16 + m];
                rmax[p][v] = fmaxf(rmax[p][v], a);
                cm = fmaxf(cm, a);
            }
            cm = fmaxf(cm, __shfl_xor(cm, 16, 32));
            if (lane < 16)
                m1part[(long)(b * ST + st0 + p) * SS + tt * 16 + lane] = cm;
        }
    }

    // row max: reduce over the 16 N-lanes, then across the 4 waves via LDS
#pragma unroll
    for (int p = 0; p < 2; ++p)
#pragma unroll
        for (int v = 0; v < 8; ++v)
#pragma unroll
            for (int msk = 1; msk < 16; msk <<= 1)
                rmax[p][v] = fmaxf(rmax[p][v], __shfl_xor(rmax[p][v], msk, 32));

    __shared__ float wm[4][32];
    if (lane == 0)
#pragma unroll
        for (int p = 0; p < 2; ++p)
#pragma unroll
            for (int v = 0; v < 8; ++v) wm[wave][p * 16 + v] = rmax[p][v];
    if (lane == 16)
#pragma unroll
        for (int p = 0; p < 2; ++p)
#pragma unroll
            for (int v = 0; v < 8; ++v) wm[wave][p * 16 + 8 + v] = rmax[p][v];
    __syncthreads();
    if (wave == 0) {
        float r = fmaxf(fmaxf(wm[0][lane], wm[1][lane]),
                        fmaxf(wm[2][lane], wm[3][lane]));
        m2[b * SS + sp * 32 + lane] = r;
    }
}

// ---------------------------------------------------------------------------
// Kernel 3: m1[b,t] = max over 100 s-tiles of partial col maxes
// ---------------------------------------------------------------------------
__global__ void m1red_kernel(const float* __restrict__ m1part,
                             float* __restrict__ m1) {
    int idx = blockIdx.x * 256 + threadIdx.x;    // < 51200
    int b = idx / SS, t = idx % SS;
    float r = -3.4e38f;
    for (int st = 0; st < ST; ++st)
        r = fmaxf(r, m1part[(long)(b * ST + st) * SS + t]);
    m1[idx] = r;
}

// ---------------------------------------------------------------------------
// Kernel 4: h = relu(bn2( bn1(concat(m1,m2)) @ w2^T + b2 ))   [32 x 2048]
// grid 2048 (one output column j), 256 threads = 32 batches x 8 lanes
// ---------------------------------------------------------------------------
__global__ void mlp_kernel(const float* __restrict__ m1,
                           const float* __restrict__ m2,
                           const float* __restrict__ g1, const float* __restrict__ be1,
                           const float* __restrict__ w2, const float* __restrict__ b2,
                           const float* __restrict__ g2, const float* __restrict__ be2,
                           float* __restrict__ h) {
    const int j = blockIdx.x;
    const int t = threadIdx.x;
    const int b = t >> 3, l8 = t & 7;
    const float* wr = w2 + (long)j * (2 * SS);
    float p = 0.f;
    for (int e = l8; e < 2 * SS; e += 8) {
        float xv = (e < SS) ? m1[b * SS + e] : m2[b * SS + e - SS];
        xv = xv * (g1[e] * RS_BN) + be1[e];
        p += wr[e] * xv;
    }
#pragma unroll
    for (int msk = 1; msk < 8; msk <<= 1) p += __shfl_xor(p, msk, 32);
    if (l8 == 0) {
        float y = (p + b2[j]) * (g2[j] * RS_BN) + be2[j];
        h[b * DFF + j] = fmaxf(y, 0.f);
    }
}

// ---------------------------------------------------------------------------
// Kernel 5: out[b] = bn3( h[b,:] @ w3 + b3 )
// ---------------------------------------------------------------------------
__global__ void head_kernel(const float* __restrict__ h,
                            const float* __restrict__ w3, const float* __restrict__ b3,
                            const float* __restrict__ g3, const float* __restrict__ be3,
                            float* __restrict__ out) {
    const int b = blockIdx.x, t = threadIdx.x;
    __shared__ float red[256];
    float p = 0.f;
    for (int j = t; j < DFF; j += 256) p += h[b * DFF + j] * w3[j];
    red[t] = p;
    __syncthreads();
    for (int s = 128; s > 0; s >>= 1) {
        if (t < s) red[t] += red[t + s];
        __syncthreads();
    }
    if (t == 0)
        out[b] = (red[0] + b3[0]) * (g3[0] * RS_BN) + be3[0];
}

// ---------------------------------------------------------------------------
extern "C" void kernel_launch(void* const* d_in, const int* in_sizes, int n_in,
                              void* d_out, int out_size, void* d_ws, size_t ws_size,
                              hipStream_t stream) {
    const float* tgt = (const float*)d_in[0];
    const float* mem = (const float*)d_in[1];
    const float* se  = (const float*)d_in[2];
    const float* w1  = (const float*)d_in[3];
    const float* b1  = (const float*)d_in[4];
    const float* g1  = (const float*)d_in[5];
    const float* be1 = (const float*)d_in[6];
    const float* w2  = (const float*)d_in[7];
    const float* b2  = (const float*)d_in[8];
    const float* g2  = (const float*)d_in[9];
    const float* be2 = (const float*)d_in[10];
    const float* w3  = (const float*)d_in[11];
    const float* b3  = (const float*)d_in[12];
    const float* g3  = (const float*)d_in[13];
    const float* be3 = (const float*)d_in[14];
    float* out = (float*)d_out;

    char* ws = (char*)d_ws;
    unsigned* Qf   = (unsigned*)(ws);                        // 13,107,200 B
    unsigned* Kf   = (unsigned*)(ws + 13107200);             // 13,107,200 B
    float* gate    = (float*)(ws + 26214400);                // 10,240,000 B
    float* m1part  = (float*)(ws + 36454400);                // 20,480,000 B
    float* m1      = (float*)(ws + 56934400);                //    204,800 B
    float* m2      = (float*)(ws + 57139200);                //    204,800 B
    float* h       = (float*)(ws + 57344000);                //    262,144 B

    qk_kernel<<<1600, 128, 0, stream>>>(tgt, mem, w1, b1, Qf, Kf);
    gate_kernel<<<(SS * SS + 255) / 256, 256, 0, stream>>>(se, gate);
    attn_kernel<<<dim3(ST / 2, BB), 128, 0, stream>>>(Qf, Kf, gate, m1part, m2);
    m1red_kernel<<<(BB * SS) / 256, 256, 0, stream>>>(m1part, m1);
    mlp_kernel<<<DFF, 256, 0, stream>>>(m1, m2, g1, be1, w2, b2, g2, be2, h);
    head_kernel<<<BB, 256, 0, stream>>>(h, w3, b3, g3, be3, out);
}
